// EnhancedConditionalUNet_56805237457393
// MI455X (gfx1250) — matrix operand order repaired and checked by reading, not compiled, not run on hardware
//
#include <hip/hip_runtime.h>
#include <hip/hip_bf16.h>
#include <math.h>

typedef __bf16 bf16_t;
typedef __attribute__((ext_vector_type(16))) __bf16 v16bf;
typedef __attribute__((ext_vector_type(8)))  float  v8f;
typedef __attribute__((ext_vector_type(4)))  int    v4i;

#define IMG 64
#define BATCH 64

// ---------------------------------------------------------------------------
// gfx1250 async global->LDS copy (GLOBAL_LOAD_ASYNC_TO_LDS_B128, ASYNCcnt),
// guarded so a toolchain without the builtin still compiles (plain 16B copy).
// Builtin signature (from hipcc diagnostic): (v4i addrspace(1)*, v4i addrspace(3)*, imm, imm)
// ---------------------------------------------------------------------------
#if defined(__has_builtin)
#if __has_builtin(__builtin_amdgcn_global_load_async_to_lds_b128)
#define ATHENA_ASYNC_LDS 1
#endif
#endif
#ifndef ATHENA_ASYNC_LDS
#define ATHENA_ASYNC_LDS 0
#endif

#define ATH_AS1 __attribute__((address_space(1)))
#define ATH_AS3 __attribute__((address_space(3)))

__device__ __forceinline__ void copy16_g2l(const bf16_t* __restrict__ g, bf16_t* l) {
#if ATHENA_ASYNC_LDS
    __builtin_amdgcn_global_load_async_to_lds_b128(
        (ATH_AS1 v4i*)((ATH_AS1 void*)(g)),
        (ATH_AS3 v4i*)((ATH_AS3 void*)(l)), 0, 0);
#else
    *(uint4*)l = *(const uint4*)g;
#endif
}
__device__ __forceinline__ void async_fence() {
#if ATHENA_ASYNC_LDS
#if __has_builtin(__builtin_amdgcn_s_wait_asynccnt)
    __builtin_amdgcn_s_wait_asynccnt(0);
#else
    asm volatile("s_wait_asynccnt 0" ::: "memory");
#endif
#endif
}

// ---------------------------------------------------------------------------
// Fragment loaders: A/B in contiguous k order (same for both, not the ISA table order)
// A 16x32 bf16: lane l -> M = l&15 ; K = (l>>4)*16 + e
// B 32x16 bf16: lane l -> N = l&15 ; K = (l>>4)*16 + e
// C/D f32:      lane l, reg r -> M = r + (l>>4)*8 ; N = l&15
// ---------------------------------------------------------------------------
__device__ __forceinline__ v16bf frag_a_from_lds(const bf16_t* As /*[64][32]*/, int m, int lane) {
    v16bf a;
    const int kb = (lane >> 4) * 16;
    const bf16_t* p = As + m * 32;
#pragma unroll
    for (int e = 0; e < 8; ++e) a[e] = p[kb + e];
#pragma unroll
    for (int e = 0; e < 8; ++e) a[8 + e] = p[kb + 8 + e];
    return a;
}
__device__ __forceinline__ v16bf frag_b_from_lds(const bf16_t* Bs /*[128][32] n-major*/, int n, int lane) {
    v16bf b;
    const int kb = (lane >> 4) * 16;
    const bf16_t* p = Bs + n * 32 + kb;
#pragma unroll
    for (int e = 0; e < 16; ++e) b[e] = p[e];
    return b;
}
// One K-chunk of the 16Mx64N wave strip: 1 A-frag, 4 B-frags, 4 WMMAs.
__device__ __forceinline__ void wmma_step(const bf16_t* As, const bf16_t* Bs,
                                          int lane, int wm, int wn, v8f acc[4]) {
    const v16bf af = frag_a_from_lds(As, wm * 16 + (lane & 15), lane);
#pragma unroll
    for (int s = 0; s < 4; ++s) {
        const v16bf bfrag = frag_b_from_lds(Bs, wn * 64 + s * 16 + (lane & 15), lane);
        acc[s] = __builtin_amdgcn_wmma_f32_16x16x32_bf16(
            false, af, false, bfrag, (short)0, acc[s], false, false);
    }
}

// ---------------------------------------------------------------------------
// prep: h0[b,c,y,x] = concat(x, label_emb[labels])[c] + pos_enc(c,y)  -> bf16
// ---------------------------------------------------------------------------
__global__ __launch_bounds__(256) void prep_kernel(
    const float* __restrict__ x, const int* __restrict__ labels,
    const float* __restrict__ lemb, bf16_t* __restrict__ h0)
{
    const int idx = blockIdx.x * 256 + threadIdx.x;     // B*4*4096
    if (idx >= BATCH * 4 * 4096) return;
    const int b = idx >> 14;
    const int rem = idx & 16383;
    const int c = rem >> 12;
    const int pix = rem & 4095;
    const int y = pix >> 6;
    float base;
    if (c < 3) base = x[(b * 3 + c) * 4096 + pix];
    else       base = lemb[labels[b] * 4096 + pix];
    const float fy = (float)y;
    float pe;
    if      (c == 0) pe = __sinf(fy);
    else if (c == 1) pe = __cosf(fy);
    else if (c == 2) pe = __sinf(fy * 0.01f);
    else             pe = __cosf(fy * 0.01f);
    h0[idx] = (bf16_t)(base + pe);
}

__global__ __launch_bounds__(256) void cvt_bf16_kernel(
    const float* __restrict__ src, bf16_t* __restrict__ dst, int n)
{
    const int i = blockIdx.x * 256 + threadIdx.x;
    if (i < n) dst[i] = (bf16_t)src[i];
}

// ---------------------------------------------------------------------------
// Generic implicit-GEMM convolution (KSZ=1 or 3, pad = KSZ==3).
// WG = 256 threads = 8 waves, tile 64(M) x 128(N), K chunks of 32 via LDS.
// Weights streamed with async global->LDS b128 when 16B-aligned (K%8==0).
// ---------------------------------------------------------------------------
template <int KSZ>
__global__ __launch_bounds__(256) void conv_gemm_kernel(
    const bf16_t* __restrict__ in, const bf16_t* __restrict__ w,
    const float* __restrict__ bias, bf16_t* __restrict__ out,
    int Bn, int Cin, int Hin, int Win,
    int Cout, int Hout, int Wout, int stride, int relu)
{
    constexpr int PAD = (KSZ == 3) ? 1 : 0;
    const int K = Cin * KSZ * KSZ;
    const int HWo = Hout * Wout;
    const int Npix = Bn * HWo;

    __shared__ __align__(16) bf16_t As[64 * 32];
    __shared__ __align__(16) bf16_t Bs[128 * 32];

    const int tid  = threadIdx.x;
    const int lane = tid & 31;
    const int wv   = tid >> 5;
    const int wm   = wv & 3;   // M sub-tile: wm*16
    const int wn   = wv >> 2;  // N strip: wn*64 (four 16-wide subtiles)

    const int nbase = blockIdx.x * 128;
    const int mbase = blockIdx.y * 64;

    v8f acc[4] = {};

    // A-fill coords: 64 rows x 32 k, 8 elems/thread
    const int fa_r = tid >> 2;
    const int fa_k = (tid & 3) * 8;
    // B-fill coords: 128 rows x 32 k, 16 elems/thread
    const int fb_r = tid >> 1;
    const int fb_k = (tid & 1) * 16;

    const int pixf = nbase + fb_r;
    const bool pv = pixf < Npix;
    int ib = 0, foy = 0, fox = 0;
    if (pv) { ib = pixf / HWo; int r2 = pixf - ib * HWo; foy = r2 / Wout; fox = r2 - foy * Wout; }

    const int gm = mbase + fa_r;
    const bool wAligned = ((K & 7) == 0);
    const int kchunks = (K + 31) >> 5;

    for (int kc = 0; kc < kchunks; ++kc) {
        const int k0 = kc << 5;
        // --- A (weights [Cout, K] row-major) ---
        if (wAligned && gm < Cout && (k0 + fa_k) < K) {
            copy16_g2l(&w[gm * K + k0 + fa_k], &As[fa_r * 32 + fa_k]);
        } else {
#pragma unroll
            for (int j = 0; j < 8; ++j) {
                const int k = k0 + fa_k + j;
                bf16_t vv = (bf16_t)0.0f;
                if (gm < Cout && k < K) vv = w[gm * K + k];
                As[fa_r * 32 + fa_k + j] = vv;
            }
        }
        // --- B (im2col, n-major [128][32]) ---
#pragma unroll
        for (int j = 0; j < 16; ++j) {
            const int k = k0 + fb_k + j;
            bf16_t vv = (bf16_t)0.0f;
            if (pv && k < K) {
                int cin, r, t2;
                if (KSZ == 1) { cin = k; r = 0; t2 = 0; }
                else { cin = k / 9; int rm = k - cin * 9; r = rm / 3; t2 = rm - r * 3; }
                const int iy = foy * stride + r - PAD;
                const int ix = fox * stride + t2 - PAD;
                if (iy >= 0 && iy < Hin && ix >= 0 && ix < Win)
                    vv = in[((ib * Cin + cin) * Hin + iy) * Win + ix];
            }
            Bs[fb_r * 32 + fb_k + j] = vv;
        }
        async_fence();
        __syncthreads();
        wmma_step(As, Bs, lane, wm, wn, acc);
        __syncthreads();
    }

    // --- epilogue: bias + optional relu, write [B, Cout, HWo] bf16 ---
    const int mrow = mbase + wm * 16 + (lane >> 4) * 8;
#pragma unroll
    for (int s = 0; s < 4; ++s) {
        const int n = nbase + wn * 64 + s * 16 + (lane & 15);
        if (n >= Npix) continue;
        const int b = n / HWo; const int rm = n - b * HWo;
#pragma unroll
        for (int r = 0; r < 8; ++r) {
            const int m = mrow + r;
            if (m >= Cout) continue;
            float vv = acc[s][r] + bias[m];
            if (relu) vv = vv > 0.f ? vv : 0.f;
            out[(size_t)(b * Cout + m) * HWo + rm] = (bf16_t)vv;
        }
    }
}

// ---------------------------------------------------------------------------
// Attention scores: s[b,i,j] = sum_c q[b,c,i]*k[b,c,j]  (K=32, one chunk)
// Tile 64(M=i) x 128(N=j) per WG.
// ---------------------------------------------------------------------------
__global__ __launch_bounds__(256) void attn_scores_kernel(
    const bf16_t* __restrict__ q, const bf16_t* __restrict__ kmat,
    bf16_t* __restrict__ scores)
{
    __shared__ __align__(16) bf16_t As[64 * 32];
    __shared__ __align__(16) bf16_t Bs[128 * 32];
    const int tid = threadIdx.x, lane = tid & 31, wv = tid >> 5;
    const int wm = wv & 3, wn = wv >> 2;
    const int nbase = blockIdx.x * 128, mbase = blockIdx.y * 64;
    const int b = blockIdx.z;
    const size_t qoff = (size_t)b * 32 * 1024;

    const int fa_r = tid >> 2, fa_k = (tid & 3) * 8;
    const int fb_r = tid >> 1, fb_k = (tid & 1) * 16;
#pragma unroll
    for (int j = 0; j < 8; ++j)
        As[fa_r * 32 + fa_k + j] = q[qoff + (size_t)(fa_k + j) * 1024 + (mbase + fa_r)];
#pragma unroll
    for (int j = 0; j < 16; ++j)
        Bs[fb_r * 32 + fb_k + j] = kmat[qoff + (size_t)(fb_k + j) * 1024 + (nbase + fb_r)];
    __syncthreads();

    v8f acc[4] = {};
    wmma_step(As, Bs, lane, wm, wn, acc);

    const size_t soff = (size_t)b << 20;
    const int mrow = mbase + wm * 16 + (lane >> 4) * 8;
#pragma unroll
    for (int s = 0; s < 4; ++s) {
        const int n = nbase + wn * 64 + s * 16 + (lane & 15);
#pragma unroll
        for (int r = 0; r < 8; ++r)
            scores[soff + (size_t)(mrow + r) * 1024 + n] = (bf16_t)acc[s][r];
    }
}

// ---------------------------------------------------------------------------
// Row softmax over last axis, in place (row = b*1024+i)
// ---------------------------------------------------------------------------
__global__ __launch_bounds__(256) void softmax_kernel(bf16_t* __restrict__ att)
{
    bf16_t* p = att + (size_t)blockIdx.x * 1024;
    const int tid = threadIdx.x;
    __shared__ float red[256];
    float vals[4];
#pragma unroll
    for (int j = 0; j < 4; ++j) vals[j] = (float)p[tid * 4 + j];
    float mx = fmaxf(fmaxf(vals[0], vals[1]), fmaxf(vals[2], vals[3]));
    red[tid] = mx; __syncthreads();
    for (int s = 128; s > 0; s >>= 1) {
        if (tid < s) red[tid] = fmaxf(red[tid], red[tid + s]);
        __syncthreads();
    }
    mx = red[0]; __syncthreads();
    float sum = 0.f;
#pragma unroll
    for (int j = 0; j < 4; ++j) { vals[j] = __expf(vals[j] - mx); sum += vals[j]; }
    red[tid] = sum; __syncthreads();
    for (int s = 128; s > 0; s >>= 1) {
        if (tid < s) red[tid] += red[tid + s];
        __syncthreads();
    }
    const float inv = 1.0f / red[0];
#pragma unroll
    for (int j = 0; j < 4; ++j) p[tid * 4 + j] = (bf16_t)(vals[j] * inv);
}

// ---------------------------------------------------------------------------
// Attention output + residual: h[b,c,i] = gamma*sum_j v[b,c,j]*att[b,i,j] + hin
// GEMM per batch: M=256(c), N=1024(i), K=1024(j); both operands contiguous in
// K -> fully async global->LDS streamed.
// ---------------------------------------------------------------------------
__global__ __launch_bounds__(256) void attn_out_kernel(
    const bf16_t* __restrict__ v, const bf16_t* __restrict__ att,
    const bf16_t* __restrict__ hin, const float* __restrict__ gamma,
    bf16_t* __restrict__ hout)
{
    __shared__ __align__(16) bf16_t As[64 * 32];
    __shared__ __align__(16) bf16_t Bs[128 * 32];
    const int tid = threadIdx.x, lane = tid & 31, wv = tid >> 5;
    const int wm = wv & 3, wn = wv >> 2;
    const int nbase = blockIdx.x * 128, mbase = blockIdx.y * 64;
    const int b = blockIdx.z;
    const size_t voff = (size_t)b * 256 * 1024;
    const size_t aoff = (size_t)b << 20;
    const int fa_r = tid >> 2, fa_k = (tid & 3) * 8;
    const int fb_r = tid >> 1, fb_k = (tid & 1) * 16;

    v8f acc[4] = {};
    for (int kc = 0; kc < 32; ++kc) {
        const int k0 = kc << 5;
        copy16_g2l(&v[voff + (size_t)(mbase + fa_r) * 1024 + k0 + fa_k],
                   &As[fa_r * 32 + fa_k]);
        copy16_g2l(&att[aoff + (size_t)(nbase + fb_r) * 1024 + k0 + fb_k],
                   &Bs[fb_r * 32 + fb_k]);
        copy16_g2l(&att[aoff + (size_t)(nbase + fb_r) * 1024 + k0 + fb_k + 8],
                   &Bs[fb_r * 32 + fb_k + 8]);
        async_fence();
        __syncthreads();
        wmma_step(As, Bs, lane, wm, wn, acc);
        __syncthreads();
    }
    const float g = gamma[0];
    const int mrow = mbase + wm * 16 + (lane >> 4) * 8;
#pragma unroll
    for (int s = 0; s < 4; ++s) {
        const int n = nbase + wn * 64 + s * 16 + (lane & 15);
#pragma unroll
        for (int r = 0; r < 8; ++r) {
            const size_t i0 = voff + (size_t)(mrow + r) * 1024 + n;
            hout[i0] = (bf16_t)(g * acc[s][r] + (float)hin[i0]);
        }
    }
}

// ---------------------------------------------------------------------------
// Pack ConvTranspose2d(128,64,k4,s2,p1) weights into 4 parity-class GEMM
// weights: weff[cls][o][i*4+dy*2+dx] = dtw[i][o][3-py-2dy][3-px-2dx]
// ---------------------------------------------------------------------------
__global__ __launch_bounds__(256) void pack_weff_kernel(
    const float* __restrict__ dtw, bf16_t* __restrict__ weff)
{
    const int idx = blockIdx.x * 256 + threadIdx.x;   // 4*64*512
    if (idx >= 4 * 64 * 512) return;
    const int cls = idx >> 15;
    const int rem = idx & 32767;
    const int o = rem >> 9;
    const int k = rem & 511;
    const int i = k >> 2, dy = (k >> 1) & 1, dx = k & 1;
    const int py = cls >> 1, px = cls & 1;
    const int ky = 3 - py - 2 * dy, kx = 3 - px - 2 * dx;
    weff[idx] = (bf16_t)dtw[((i * 64 + o) * 4 + ky) * 4 + kx];
}

// ---------------------------------------------------------------------------
// Transposed conv as implicit GEMM per parity class (blockIdx.z = class).
// M=64, N=B*32*32=65536 per class, K=512. Tile 64x128.
// ---------------------------------------------------------------------------
__global__ __launch_bounds__(256) void deconv_gemm_kernel(
    const bf16_t* __restrict__ in /*[B,128,32,32]*/,
    const bf16_t* __restrict__ weff /*[4][64][512]*/,
    const float* __restrict__ bias, bf16_t* __restrict__ out /*[B,64,64,64]*/)
{
    __shared__ __align__(16) bf16_t As[64 * 32];
    __shared__ __align__(16) bf16_t Bs[128 * 32];
    const int cls = blockIdx.z;
    const int py = cls >> 1, px = cls & 1;
    const int tid = threadIdx.x, lane = tid & 31, wv = tid >> 5;
    const int wm = wv & 3, wn = wv >> 2;
    const int nbase = blockIdx.x * 128;
    const int fa_r = tid >> 2, fa_k = (tid & 3) * 8;
    const int fb_r = tid >> 1, fb_k = (tid & 1) * 16;

    const int pixf = nbase + fb_r;
    const int ib = pixf >> 10;
    const int prem = pixf & 1023;
    const int ry = prem >> 5, rx = prem & 31;
    const bf16_t* wcls = weff + (size_t)cls * 64 * 512;

    v8f acc[4] = {};
    for (int kc = 0; kc < 16; ++kc) {
        const int k0 = kc << 5;
        copy16_g2l(&wcls[fa_r * 512 + k0 + fa_k], &As[fa_r * 32 + fa_k]);
#pragma unroll
        for (int j = 0; j < 16; ++j) {
            const int k = k0 + fb_k + j;
            const int i = k >> 2, dy = (k >> 1) & 1, dx = k & 1;
            const int iy = ry - 1 + py + dy;
            const int ix = rx - 1 + px + dx;
            bf16_t vv = (bf16_t)0.0f;
            if (iy >= 0 && iy < 32 && ix >= 0 && ix < 32)
                vv = in[((ib * 128 + i) * 32 + iy) * 32 + ix];
            Bs[fb_r * 32 + fb_k + j] = vv;
        }
        async_fence();
        __syncthreads();
        wmma_step(As, Bs, lane, wm, wn, acc);
        __syncthreads();
    }

    const int mrow = wm * 16 + (lane >> 4) * 8;
#pragma unroll
    for (int s = 0; s < 4; ++s) {
        const int n = nbase + wn * 64 + s * 16 + (lane & 15);
        const int b = n >> 10;
        const int rem2 = n & 1023;
        const int y = 2 * (rem2 >> 5) + py;
        const int x = 2 * (rem2 & 31) + px;
#pragma unroll
        for (int r = 0; r < 8; ++r) {
            const int m = mrow + r;
            float vv = acc[s][r] + bias[m];
            vv = vv > 0.f ? vv : 0.f;
            out[((size_t)(b * 64 + m) * 64 + y) * 64 + x] = (bf16_t)vv;
        }
    }
}

// ---------------------------------------------------------------------------
// Final conv 64->3, 3x3, pad 1, tanh, fp32 output (small: direct VALU)
// ---------------------------------------------------------------------------
__global__ __launch_bounds__(256) void final_conv_kernel(
    const bf16_t* __restrict__ h /*[B,64,64,64]*/,
    const float* __restrict__ wgt /*[3,64,3,3]*/,
    const float* __restrict__ bias, float* __restrict__ out)
{
    const int idx = blockIdx.x * 256 + threadIdx.x;   // B*4096
    if (idx >= BATCH * 4096) return;
    const int b = idx >> 12;
    const int pix = idx & 4095;
    const int y = pix >> 6, x = pix & 63;
    float acc[3] = { bias[0], bias[1], bias[2] };
    for (int i = 0; i < 64; ++i) {
#pragma unroll
        for (int r = 0; r < 3; ++r) {
            const int iy = y + r - 1;
            if (iy < 0 || iy > 63) continue;
#pragma unroll
            for (int t = 0; t < 3; ++t) {
                const int ix = x + t - 1;
                if (ix < 0 || ix > 63) continue;
                const float hv = (float)h[((size_t)(b * 64 + i) * 64 + iy) * 64 + ix];
#pragma unroll
                for (int oc = 0; oc < 3; ++oc)
                    acc[oc] += hv * wgt[((oc * 64 + i) * 3 + r) * 3 + t];
            }
        }
    }
#pragma unroll
    for (int oc = 0; oc < 3; ++oc)
        out[(size_t)(b * 3 + oc) * 4096 + pix] = tanhf(acc[oc]);
}

// ---------------------------------------------------------------------------
// Host-side orchestration
// ---------------------------------------------------------------------------
static inline void* wsp(void* ws, size_t& off, size_t bytes) {
    void* p = (char*)ws + off;
    off += (bytes + 255) & ~(size_t)255;
    return p;
}

extern "C" void kernel_launch(void* const* d_in, const int* in_sizes, int n_in,
                              void* d_out, int out_size, void* d_ws, size_t ws_size,
                              hipStream_t stream) {
    (void)in_sizes; (void)n_in; (void)out_size; (void)ws_size;
    const float* x        = (const float*)d_in[0];
    const int*   labels   = (const int*)  d_in[2];
    const float* lemb     = (const float*)d_in[3];
    const float* ew1      = (const float*)d_in[4];
    const float* eb1      = (const float*)d_in[5];
    const float* ew2      = (const float*)d_in[6];
    const float* eb2      = (const float*)d_in[7];
    const float* bw1      = (const float*)d_in[8];
    const float* bb1      = (const float*)d_in[9];
    const float* qw       = (const float*)d_in[10];
    const float* qbias    = (const float*)d_in[11];
    const float* kw       = (const float*)d_in[12];
    const float* kbias    = (const float*)d_in[13];
    const float* vw       = (const float*)d_in[14];
    const float* vbias    = (const float*)d_in[15];
    const float* gamma    = (const float*)d_in[16];
    const float* bw2      = (const float*)d_in[17];
    const float* bb2      = (const float*)d_in[18];
    const float* dtw      = (const float*)d_in[19];
    const float* dtb      = (const float*)d_in[20];
    const float* dcw      = (const float*)d_in[21];
    const float* dcb      = (const float*)d_in[22];

    size_t off = 0;
    bf16_t* h0   = (bf16_t*)wsp(d_ws, off, (size_t)BATCH * 4   * 4096 * 2);
    bf16_t* w1b  = (bf16_t*)wsp(d_ws, off, (size_t)64  * 36   * 2);
    bf16_t* h1   = (bf16_t*)wsp(d_ws, off, (size_t)BATCH * 64  * 4096 * 2);
    bf16_t* w2b  = (bf16_t*)wsp(d_ws, off, (size_t)128 * 576  * 2);
    bf16_t* h2   = (bf16_t*)wsp(d_ws, off, (size_t)BATCH * 128 * 1024 * 2);
    bf16_t* wb1b = (bf16_t*)wsp(d_ws, off, (size_t)256 * 1152 * 2);
    bf16_t* h3   = (bf16_t*)wsp(d_ws, off, (size_t)BATCH * 256 * 1024 * 2);
    bf16_t* wqb  = (bf16_t*)wsp(d_ws, off, (size_t)32  * 256  * 2);
    bf16_t* wkb  = (bf16_t*)wsp(d_ws, off, (size_t)32  * 256  * 2);
    bf16_t* wvb  = (bf16_t*)wsp(d_ws, off, (size_t)256 * 256  * 2);
    bf16_t* qbuf = (bf16_t*)wsp(d_ws, off, (size_t)BATCH * 32  * 1024 * 2);
    bf16_t* kbuf = (bf16_t*)wsp(d_ws, off, (size_t)BATCH * 32  * 1024 * 2);
    bf16_t* vbuf = (bf16_t*)wsp(d_ws, off, (size_t)BATCH * 256 * 1024 * 2);
    bf16_t* att  = (bf16_t*)wsp(d_ws, off, (size_t)BATCH * 1024 * 1024 * 2);
    bf16_t* hatt = (bf16_t*)wsp(d_ws, off, (size_t)BATCH * 256 * 1024 * 2);
    bf16_t* wb2b = (bf16_t*)wsp(d_ws, off, (size_t)128 * 2304 * 2);
    bf16_t* h4   = (bf16_t*)wsp(d_ws, off, (size_t)BATCH * 128 * 1024 * 2);
    bf16_t* weff = (bf16_t*)wsp(d_ws, off, (size_t)4 * 64 * 512 * 2);
    bf16_t* h5   = (bf16_t*)wsp(d_ws, off, (size_t)BATCH * 64  * 4096 * 2);

    // --- weight conversions + input prep ---
    auto cvt = [&](const float* s, bf16_t* d, int n) {
        cvt_bf16_kernel<<<(n + 255) / 256, 256, 0, stream>>>(s, d, n);
    };
    cvt(ew1, w1b,  64 * 36);
    cvt(ew2, w2b,  128 * 576);
    cvt(bw1, wb1b, 256 * 1152);
    cvt(qw,  wqb,  32 * 256);
    cvt(kw,  wkb,  32 * 256);
    cvt(vw,  wvb,  256 * 256);
    cvt(bw2, wb2b, 128 * 2304);
    pack_weff_kernel<<<(4 * 64 * 512 + 255) / 256, 256, 0, stream>>>(dtw, weff);
    prep_kernel<<<(BATCH * 4 * 4096 + 255) / 256, 256, 0, stream>>>(x, labels, lemb, h0);

    // --- encoder ---
    conv_gemm_kernel<3><<<dim3(2048, 1), 256, 0, stream>>>(
        h0, w1b, eb1, h1, BATCH, 4, 64, 64, 64, 64, 64, 1, 1);
    conv_gemm_kernel<3><<<dim3(512, 2), 256, 0, stream>>>(
        h1, w2b, eb2, h2, BATCH, 64, 64, 64, 128, 32, 32, 2, 1);

    // --- bottleneck conv ---
    conv_gemm_kernel<3><<<dim3(512, 4), 256, 0, stream>>>(
        h2, wb1b, bb1, h3, BATCH, 128, 32, 32, 256, 32, 32, 1, 1);

    // --- q, k, v (1x1 convs, no relu) ---
    conv_gemm_kernel<1><<<dim3(512, 1), 256, 0, stream>>>(
        h3, wqb, qbias, qbuf, BATCH, 256, 32, 32, 32, 32, 32, 1, 0);
    conv_gemm_kernel<1><<<dim3(512, 1), 256, 0, stream>>>(
        h3, wkb, kbias, kbuf, BATCH, 256, 32, 32, 32, 32, 32, 1, 0);
    conv_gemm_kernel<1><<<dim3(512, 4), 256, 0, stream>>>(
        h3, wvb, vbias, vbuf, BATCH, 256, 32, 32, 256, 32, 32, 1, 0);

    // --- attention ---
    attn_scores_kernel<<<dim3(8, 16, BATCH), 256, 0, stream>>>(qbuf, kbuf, att);
    softmax_kernel<<<BATCH * 1024, 256, 0, stream>>>(att);
    attn_out_kernel<<<dim3(8, 4, BATCH), 256, 0, stream>>>(vbuf, att, h3, gamma, hatt);

    // --- post-attention conv ---
    conv_gemm_kernel<3><<<dim3(512, 2), 256, 0, stream>>>(
        hatt, wb2b, bb2, h4, BATCH, 256, 32, 32, 128, 32, 32, 1, 1);

    // --- decoder: transposed conv as 4 parity-class GEMMs ---
    deconv_gemm_kernel<<<dim3(512, 1, 4), 256, 0, stream>>>(h4, weff, dtb, h5);

    // --- final conv + tanh -> fp32 output ---
    final_conv_kernel<<<(BATCH * 4096 + 255) / 256, 256, 0, stream>>>(
        h5, dcw, dcb, (float*)d_out);
}
